// VJEPA2EncoderWrapper_63960652972271
// MI455X (gfx1250) — compile-verified, matrix operand order checked by reading, not run on hardware
//
#include <hip/hip_runtime.h>
#include <hip/hip_bf16.h>
#include <math.h>

// ---------------------------------------------------------------------------
// VJEPA2 encoder forward for MI455X (gfx1250, wave32, WMMA).
// Mixed precision: f16 operands, f32 accumulation via v_wmma_f32_16x16x32_f16.
// ---------------------------------------------------------------------------

typedef __attribute__((ext_vector_type(16))) _Float16 v16h;
typedef __attribute__((ext_vector_type(8)))  float    v8f;

#define HID   1024
#define NT    1568   // tokens per batch
#define MTOT  3136   // B * NT
#define MLPD  4096
#define NHEAD 16
#define HD    64
#define KPATCH 1536  // 3*2*16*16

// ---------------------------------------------------------------------------
// WMMA helper
// ---------------------------------------------------------------------------
__device__ __forceinline__ v8f wmma32(v16h a, v16h b, v8f c) {
  // D = A(16x32 f16) * B(32x16 f16) + C(16x16 f32)
  return __builtin_amdgcn_wmma_f32_16x16x32_f16(
      /*neg_a=*/false, a, /*neg_b=*/false, b,
      /*c_mod=*/(short)0, c, /*reuse_a=*/false, /*reuse_b=*/false);
}

// Load a 16x32 f16 fragment (A layout; also used for B since B's column n is
// a K-contiguous row of the row-major [N][K] weight matrix).
// Layout (ISA 7.12.2): lane&15 = row; lanes<16 hold K {k0..k0+7, k0+16..k0+23},
// lanes>=16 hold K {k0+8..k0+15, k0+24..k0+31}.
__device__ __forceinline__ v16h load_frag(const _Float16* base, int ld,
                                          int row0, int k0, int lane) {
  const _Float16* p = base + (size_t)(row0 + (lane & 15)) * ld
                           + (k0 + ((lane & 16) ? 8 : 0));
  v16h f;
  int4* fp = reinterpret_cast<int4*>(&f);
  fp[0] = *reinterpret_cast<const int4*>(p);        // halves 0..7
  fp[1] = *reinterpret_cast<const int4*>(p + 16);   // halves 8..15
  return f;
}

// Fused-im2col A fragment for the patch-embed GEMM.
// K index = c*512 + t*256 + ph*16 + pw  (matches patch_w OIDHW row layout).
// 8-element K groups never cross a pw row -> two aligned float4 loads.
__device__ __forceinline__ v16h load_frag_patch(const float* px, int row0,
                                                int k0, int lane) {
  int t = row0 + (lane & 15);
  int b = t / NT;           int n  = t - b * NT;
  int d = n / 196;          int rm = n - d * 196;
  int gh = rm / 14;         int gw = rm - gh * 14;
  int kk = k0 + ((lane & 16) ? 8 : 0);
  v16h f;
#pragma unroll
  for (int g = 0; g < 2; ++g) {
    int k  = kk + 16 * g;
    int c  = k >> 9;
    int tt = (k >> 8) & 1;
    int ph = (k >> 4) & 15;
    int pw = k & 15;
    const float* p = px +
        (((size_t)((b * 16 + d * 2 + tt) * 3 + c) * 224 + gh * 16 + ph) * 224
         + gw * 16 + pw);
    float4 x0 = *reinterpret_cast<const float4*>(p);
    float4 x1 = *reinterpret_cast<const float4*>(p + 4);
    f[8*g+0] = (_Float16)x0.x; f[8*g+1] = (_Float16)x0.y;
    f[8*g+2] = (_Float16)x0.z; f[8*g+3] = (_Float16)x0.w;
    f[8*g+4] = (_Float16)x1.x; f[8*g+5] = (_Float16)x1.y;
    f[8*g+6] = (_Float16)x1.z; f[8*g+7] = (_Float16)x1.w;
  }
  return f;
}

// ---------------------------------------------------------------------------
// f32 -> f16 conversion (weights)
// ---------------------------------------------------------------------------
__global__ void cvt_f16_kernel(const float* __restrict__ s,
                               _Float16* __restrict__ d, int n) {
  int i = blockIdx.x * 256 + threadIdx.x;
  if (i < n) d[i] = (_Float16)s[i];
}

// ---------------------------------------------------------------------------
// Generic GEMM: C[m,n] = sum_k A[m,k]*B[n,k] + bias[n]
// A: f16 [M][K], B: f16 [N][K] (weights are stored (out,in) -> row n is col n)
// Each wave computes a 64x64 tile as 4x4 WMMA tiles. 8 waves / block.
// epi: 0 = store f32, 1 = f32 += (residual), 3 = gelu -> f16
// ---------------------------------------------------------------------------
__global__ __launch_bounds__(256)
void gemm_kernel(const _Float16* __restrict__ A, const _Float16* __restrict__ B,
                 const float* __restrict__ bias, int M, int N, int K, int epi,
                 float* __restrict__ Cf, _Float16* __restrict__ Ch) {
  int lane = threadIdx.x & 31;
  int wid  = threadIdx.x >> 5;
  int id   = blockIdx.x * 8 + wid;
  int tn   = N >> 6;
  int total = (M >> 6) * tn;
  if (id >= total) return;
  int row0 = (id / tn) * 64;
  int col0 = (id % tn) * 64;

  v8f acc[4][4] = {};
  for (int k0 = 0; k0 < K; k0 += 32) {
    v16h af[4], bf[4];
#pragma unroll
    for (int i = 0; i < 4; ++i) af[i] = load_frag(A, K, row0 + 16*i, k0, lane);
#pragma unroll
    for (int j = 0; j < 4; ++j) bf[j] = load_frag(B, K, col0 + 16*j, k0, lane);
#pragma unroll
    for (int i = 0; i < 4; ++i)
#pragma unroll
      for (int j = 0; j < 4; ++j) acc[i][j] = wmma32(af[i], bf[j], acc[i][j]);
  }

  int half = (lane >> 4) & 1;
  int cl   = lane & 15;
#pragma unroll
  for (int i = 0; i < 4; ++i) {
#pragma unroll
    for (int j = 0; j < 4; ++j) {
#pragma unroll
      for (int r = 0; r < 8; ++r) {
        int row = row0 + 16*i + r + 8*half;
        int col = col0 + 16*j + cl;
        float v = acc[i][j][r] + bias[col];
        size_t o = (size_t)row * N + col;
        if (epi == 0)      Cf[o] = v;
        else if (epi == 1) Cf[o] += v;
        else /* gelu */    Ch[o] = (_Float16)(0.5f * v * (1.0f + erff(v * 0.70710678118654752f)));
      }
    }
  }
}

// ---------------------------------------------------------------------------
// Patch-embed GEMM: M=3136, N=1024, K=1536, A gathered from video pixels.
// ---------------------------------------------------------------------------
__global__ __launch_bounds__(256)
void patch_gemm_kernel(const float* __restrict__ px, const _Float16* __restrict__ W,
                       const float* __restrict__ bias, float* __restrict__ C) {
  int lane = threadIdx.x & 31;
  int wid  = threadIdx.x >> 5;
  int id   = blockIdx.x * 8 + wid;              // 784 tiles
  if (id >= (MTOT/64) * (HID/64)) return;
  int tn = HID >> 6;
  int row0 = (id / tn) * 64;
  int col0 = (id % tn) * 64;

  v8f acc[4][4] = {};
  for (int k0 = 0; k0 < KPATCH; k0 += 32) {
    v16h af[4], bf[4];
#pragma unroll
    for (int i = 0; i < 4; ++i) af[i] = load_frag_patch(px, row0 + 16*i, k0, lane);
#pragma unroll
    for (int j = 0; j < 4; ++j) bf[j] = load_frag(W, KPATCH, col0 + 16*j, k0, lane);
#pragma unroll
    for (int i = 0; i < 4; ++i)
#pragma unroll
      for (int j = 0; j < 4; ++j) acc[i][j] = wmma32(af[i], bf[j], acc[i][j]);
  }
  int half = (lane >> 4) & 1, cl = lane & 15;
#pragma unroll
  for (int i = 0; i < 4; ++i)
#pragma unroll
    for (int j = 0; j < 4; ++j)
#pragma unroll
      for (int r = 0; r < 8; ++r) {
        int row = row0 + 16*i + r + 8*half;
        int col = col0 + 16*j + cl;
        C[(size_t)row * HID + col] = acc[i][j][r] + bias[col];
      }
}

// ---------------------------------------------------------------------------
// LayerNorm over 1024 cols. One block per row. Output f16 (yh) or f32 (yf).
// ---------------------------------------------------------------------------
__global__ __launch_bounds__(256)
void ln_kernel(const float* __restrict__ x, const float* __restrict__ g,
               const float* __restrict__ b, _Float16* __restrict__ yh,
               float* __restrict__ yf) {
  __shared__ float red[256];
  int row = blockIdx.x;
  const float* xr = x + (size_t)row * HID;
  float s = 0.f;
  for (int c = threadIdx.x; c < HID; c += 256) s += xr[c];
  red[threadIdx.x] = s; __syncthreads();
  for (int st = 128; st > 0; st >>= 1) {
    if (threadIdx.x < st) red[threadIdx.x] += red[threadIdx.x + st];
    __syncthreads();
  }
  float mu = red[0] * (1.0f / HID);
  __syncthreads();
  float v = 0.f;
  for (int c = threadIdx.x; c < HID; c += 256) {
    float d = xr[c] - mu; v += d * d;
  }
  red[threadIdx.x] = v; __syncthreads();
  for (int st = 128; st > 0; st >>= 1) {
    if (threadIdx.x < st) red[threadIdx.x] += red[threadIdx.x + st];
    __syncthreads();
  }
  float inv = rsqrtf(red[0] * (1.0f / HID) + 1e-6f);
  for (int c = threadIdx.x; c < HID; c += 256) {
    float o = (xr[c] - mu) * inv * g[c] + b[c];
    if (yh) yh[(size_t)row * HID + c] = (_Float16)o;
    else    yf[(size_t)row * HID + c] = o;
  }
}

// ---------------------------------------------------------------------------
// RoPE + head arrange: f32 staging [3136][1024] -> f16 [B,H,N,64]
// Exact VJEPA2 quirk: tiled sin/cos (index j%10) * interleaved rotate-half.
// ---------------------------------------------------------------------------
__global__ void rope_arrange_kernel(const float* __restrict__ st,
                                    _Float16* __restrict__ dst) {
  int idx = blockIdx.x * 256 + threadIdx.x;      // (token, head)
  if (idx >= MTOT * NHEAD) return;
  int t = idx >> 4, hh = idx & 15;
  int b = t / NT, n = t - b * NT;
  int d = n / 196, rm = n - d * 196, hp = rm / 14, wp = rm - hp * 14;
  const float* src = st + (size_t)t * HID + hh * HD;
  _Float16* out = dst + ((size_t)((b * NHEAD + hh) * NT) + n) * HD;
  float pos[3] = {(float)d, (float)hp, (float)wp};
#pragma unroll
  for (int ax = 0; ax < 3; ++ax) {
    const float* xx = src + ax * 20;
    _Float16* oo = out + ax * 20;
    float fr[10];
#pragma unroll
    for (int j = 0; j < 10; ++j)
      fr[j] = pos[ax] * __powf(10000.f, -(float)j * 0.1f);
#pragma unroll
    for (int i = 0; i < 10; ++i) {
      float x0 = xx[2*i], x1 = xx[2*i+1];
      float s0, c0, s1, c1;
      __sincosf(fr[(2*i)   % 10], &s0, &c0);
      __sincosf(fr[(2*i+1) % 10], &s1, &c1);
      oo[2*i]   = (_Float16)(x0 * c0 - x1 * s0);
      oo[2*i+1] = (_Float16)(x1 * c1 + x0 * s1);
    }
  }
  for (int dd = 60; dd < 64; ++dd) out[dd] = (_Float16)src[dd];
}

// V arrange (transposed): f32 staging [3136][1024] -> f16 [B,H,64,N]
__global__ void v_arrange_kernel(const float* __restrict__ st,
                                 _Float16* __restrict__ vt) {
  int idx = blockIdx.x * 256 + threadIdx.x;
  if (idx >= MTOT * HID) return;
  int t = idx >> 10, col = idx & 1023;
  int hh = col >> 6, dd = col & 63;
  int b = t / NT, n = t - b * NT;
  vt[((size_t)((b * NHEAD + hh) * HD) + dd) * NT + n] =
      (_Float16)st[(size_t)t * HID + col];
}

// ---------------------------------------------------------------------------
// Flash attention. q,k: f16 [B*H][N][64]; vt: f16 [B*H][64][N].
// Grid: x = ceil(98/4) row-tile groups, y = B*H. 4 waves/block, 1 wave per
// 16-row tile. Online softmax; P tile goes C-layout -> LDS -> A-layout.
// ---------------------------------------------------------------------------
__global__ __launch_bounds__(128)
void attn_kernel(const _Float16* __restrict__ q, const _Float16* __restrict__ k,
                 const _Float16* __restrict__ vt, _Float16* __restrict__ out) {
  __shared__ __align__(16) _Float16 lds[4][16 * 32];
  int lane = threadIdx.x & 31;
  int wid  = threadIdx.x >> 5;
  int tile = blockIdx.x * 4 + wid;
  int bh   = blockIdx.y;
  if (tile >= NT / 16) return;
  int b = bh >> 4, hh = bh & 15;
  const _Float16* qb = q  + (size_t)bh * NT * HD;
  const _Float16* kb = k  + (size_t)bh * NT * HD;
  const _Float16* vb = vt + (size_t)bh * HD * NT;
  int row0 = tile * 16;

  v16h qf0 = load_frag(qb, HD, row0, 0,  lane);
  v16h qf1 = load_frag(qb, HD, row0, 32, lane);

  v8f o0 = {}, o1 = {}, o2 = {}, o3 = {};
  float mrun[8], lrun[8];
#pragma unroll
  for (int r = 0; r < 8; ++r) { mrun[r] = -1e30f; lrun[r] = 0.f; }

  _Float16* L = lds[wid];
  int half = (lane >> 4) & 1;
  int col  = lane & 15;

  for (int kt = 0; kt < NT / 32; ++kt) {
    int j0 = kt * 32, j1 = j0 + 16;
    v8f s0 = {}, s1 = {};
    s0 = wmma32(qf0, load_frag(kb, HD, j0, 0,  lane), s0);
    s0 = wmma32(qf1, load_frag(kb, HD, j0, 32, lane), s0);
    s1 = wmma32(qf0, load_frag(kb, HD, j1, 0,  lane), s1);
    s1 = wmma32(qf1, load_frag(kb, HD, j1, 32, lane), s1);

#pragma unroll
    for (int r = 0; r < 8; ++r) {
      float a0 = s0[r] * 0.125f;     // HEAD^-0.5
      float a1 = s1[r] * 0.125f;
      float mx = fmaxf(a0, a1);
#pragma unroll
      for (int off = 1; off < 16; off <<= 1) mx = fmaxf(mx, __shfl_xor(mx, off, 32));
      float mnew  = fmaxf(mrun[r], mx);
      float alpha = __expf(mrun[r] - mnew);
      float p0 = __expf(a0 - mnew);
      float p1 = __expf(a1 - mnew);
      float rs = p0 + p1;
#pragma unroll
      for (int off = 1; off < 16; off <<= 1) rs += __shfl_xor(rs, off, 32);
      lrun[r] = lrun[r] * alpha + rs;
      mrun[r] = mnew;
      o0[r] *= alpha; o1[r] *= alpha; o2[r] *= alpha; o3[r] *= alpha;
      s0[r] = p0; s1[r] = p1;
    }

    // C-layout -> row-major LDS [16][32] -> A-layout fragment.
#pragma unroll
    for (int r = 0; r < 8; ++r) {
      int row = r + 8 * half;
      L[row * 32 + col]      = (_Float16)s0[r];
      L[row * 32 + 16 + col] = (_Float16)s1[r];
    }
    asm volatile("s_wait_dscnt 0" ::: "memory");   // CDNA5 split DS counter
    v16h pf = load_frag(L, 32, 0, 0, lane);

    o0 = wmma32(pf, load_frag(vb, NT, 0,  j0, lane), o0);
    o1 = wmma32(pf, load_frag(vb, NT, 16, j0, lane), o1);
    o2 = wmma32(pf, load_frag(vb, NT, 32, j0, lane), o2);
    o3 = wmma32(pf, load_frag(vb, NT, 48, j0, lane), o3);
  }

#pragma unroll
  for (int r = 0; r < 8; ++r) {
    int row = row0 + r + 8 * half;
    float inv = 1.0f / lrun[r];
    size_t base = ((size_t)(b * NT + row)) * HID + hh * HD;
    out[base +      col] = (_Float16)(o0[r] * inv);
    out[base + 16 + col] = (_Float16)(o1[r] * inv);
    out[base + 32 + col] = (_Float16)(o2[r] * inv);
    out[base + 48 + col] = (_Float16)(o3[r] * inv);
  }
}

// ---------------------------------------------------------------------------
// Host orchestration
// ---------------------------------------------------------------------------
extern "C" void kernel_launch(void* const* d_in, const int* in_sizes, int n_in,
                              void* d_out, int out_size, void* d_ws, size_t ws_size,
                              hipStream_t stream) {
  const float* px      = (const float*)d_in[0];
  const float* patch_w = (const float*)d_in[1];
  const float* patch_b = (const float*)d_in[2];
  const float* ln1_g   = (const float*)d_in[3];
  const float* ln1_b   = (const float*)d_in[4];
  const float* qw      = (const float*)d_in[5];
  const float* qbias   = (const float*)d_in[6];
  const float* kw      = (const float*)d_in[7];
  const float* kbias   = (const float*)d_in[8];
  const float* vw      = (const float*)d_in[9];
  const float* vbias   = (const float*)d_in[10];
  const float* pw      = (const float*)d_in[11];
  const float* pbias   = (const float*)d_in[12];
  const float* ln2_g   = (const float*)d_in[13];
  const float* ln2_b   = (const float*)d_in[14];
  const float* fc1w    = (const float*)d_in[15];
  const float* fc1b    = (const float*)d_in[16];
  const float* fc2w    = (const float*)d_in[17];
  const float* fc2b    = (const float*)d_in[18];
  const float* lnf_g   = (const float*)d_in[19];
  const float* lnf_b   = (const float*)d_in[20];

  char* base = (char*)d_ws;
  size_t off = 0;
  auto alloc = [&](size_t bytes) -> void* {
    void* p = base + off;
    off = (off + bytes + 255) & ~(size_t)255;
    return p;
  };

  _Float16* wp16  = (_Float16*)alloc((size_t)HID * KPATCH * 2);
  _Float16* qw16  = (_Float16*)alloc((size_t)4 * HID * HID * 2);
  _Float16* kw16  = (_Float16*)alloc((size_t)4 * HID * HID * 2);
  _Float16* vw16  = (_Float16*)alloc((size_t)4 * HID * HID * 2);
  _Float16* pw16  = (_Float16*)alloc((size_t)4 * HID * HID * 2);
  _Float16* f1w16 = (_Float16*)alloc((size_t)4 * MLPD * HID * 2);
  _Float16* f2w16 = (_Float16*)alloc((size_t)4 * HID * MLPD * 2);
  float*    hbuf  = (float*)   alloc((size_t)MTOT * HID * 4);
  float*    stage = (float*)   alloc((size_t)MTOT * HID * 4);
  _Float16* xn    = (_Float16*)alloc((size_t)MTOT * HID * 2);
  _Float16* qf    = (_Float16*)alloc((size_t)MTOT * HID * 2);
  _Float16* kf    = (_Float16*)alloc((size_t)MTOT * HID * 2);
  _Float16* vtf   = (_Float16*)alloc((size_t)MTOT * HID * 2);
  _Float16* attn  = (_Float16*)alloc((size_t)MTOT * HID * 2);
  _Float16* mid   = (_Float16*)alloc((size_t)MTOT * MLPD * 2);

  auto cvt = [&](const float* s, _Float16* d, int n) {
    cvt_f16_kernel<<<(n + 255) / 256, 256, 0, stream>>>(s, d, n);
  };
  cvt(patch_w, wp16, HID * KPATCH);
  cvt(qw, qw16, 4 * HID * HID);
  cvt(kw, kw16, 4 * HID * HID);
  cvt(vw, vw16, 4 * HID * HID);
  cvt(pw, pw16, 4 * HID * HID);
  cvt(fc1w, f1w16, 4 * MLPD * HID);
  cvt(fc2w, f2w16, 4 * HID * MLPD);

  // Patch embed: h = im2col(pixels) @ patch_w^T + patch_b
  patch_gemm_kernel<<<(MTOT/64)*(HID/64)/8, 256, 0, stream>>>(px, wp16, patch_b, hbuf);

  int g1024 = ((MTOT/64) * (HID/64)  + 7) / 8;   // 98
  int g4096 = ((MTOT/64) * (MLPD/64) + 7) / 8;   // 392

  for (int i = 0; i < 4; ++i) {
    size_t wo  = (size_t)i * HID * HID;
    size_t w1o = (size_t)i * MLPD * HID;

    // ---- attention block ----
    ln_kernel<<<MTOT, 256, 0, stream>>>(hbuf, ln1_g + i*HID, ln1_b + i*HID, xn, nullptr);

    gemm_kernel<<<g1024, 256, 0, stream>>>(xn, qw16 + wo, qbias + i*HID,
                                           MTOT, HID, HID, 0, stage, nullptr);
    rope_arrange_kernel<<<(MTOT*NHEAD + 255)/256, 256, 0, stream>>>(stage, qf);

    gemm_kernel<<<g1024, 256, 0, stream>>>(xn, kw16 + wo, kbias + i*HID,
                                           MTOT, HID, HID, 0, stage, nullptr);
    rope_arrange_kernel<<<(MTOT*NHEAD + 255)/256, 256, 0, stream>>>(stage, kf);

    gemm_kernel<<<g1024, 256, 0, stream>>>(xn, vw16 + wo, vbias + i*HID,
                                           MTOT, HID, HID, 0, stage, nullptr);
    v_arrange_kernel<<<((size_t)MTOT*HID + 255)/256, 256, 0, stream>>>(stage, vtf);

    attn_kernel<<<dim3((NT/16 + 3)/4, 2*NHEAD), 128, 0, stream>>>(qf, kf, vtf, attn);

    gemm_kernel<<<g1024, 256, 0, stream>>>(attn, pw16 + wo, pbias + i*HID,
                                           MTOT, HID, HID, 1, hbuf, nullptr);

    // ---- MLP block ----
    ln_kernel<<<MTOT, 256, 0, stream>>>(hbuf, ln2_g + i*HID, ln2_b + i*HID, xn, nullptr);
    gemm_kernel<<<g4096, 256, 0, stream>>>(xn, f1w16 + w1o, fc1b + i*MLPD,
                                           MTOT, MLPD, HID, 3, nullptr, mid);
    gemm_kernel<<<g1024, 256, 0, stream>>>(mid, f2w16 + w1o, fc2b + i*HID,
                                           MTOT, HID, MLPD, 1, hbuf, nullptr);
  }

  // Final LN -> fp32 output (B, N, 1024)
  ln_kernel<<<MTOT, 256, 0, stream>>>(hbuf, lnf_g, lnf_b, nullptr, (float*)d_out);
}
